// DecoderCell_22162031247572
// MI455X (gfx1250) — matrix operand order repaired and checked by reading, not compiled, not run on hardware
//
#include <hip/hip_runtime.h>
#include <hip/hip_bf16.h>
#include <math.h>

// ---------------------------------------------------------------------------
// Problem constants
// ---------------------------------------------------------------------------
#define BS       64
#define N        4096
#define NS       4095
#define D        128
#define NHEADS   8
#define DH       16
#define CHUNK    256          // rows per workgroup tile
#define NCHUNK   16           // N / CHUNK
#define MAX_NUM  99999.0f
#define TINYV    0.001f
#define NEG_INF  -10000000000.0f
#define CLIPV    10.0f
#define INV_SQRT_DH  0.25f            // 1/sqrt(16)
#define INV_SQRT_D   0.08838834764831845f  // 1/sqrt(128)

// ---------------------------------------------------------------------------
// Workspace layout (floats)
// ---------------------------------------------------------------------------
#define OFF_Q1     0                        // 64*128
#define OFF_ATTP   8192                     // 64*16*8*18 = 147456
#define OFF_MM     155648                   // 64*16*2   (min,max per chunk)
#define OFF_SNORM  157696                   // 64*2      (mn, ds per batch)
#define OFF_Q2     157824                   // 64*128
#define OFF_PSUM   166016                   // 64*16
// total 167040 floats = 668160 bytes

// ---------------------------------------------------------------------------
// WMMA types (gfx1250, wave32)
// ---------------------------------------------------------------------------
typedef __attribute__((ext_vector_type(16))) __bf16 v16bf;
typedef __attribute__((ext_vector_type(8)))  __bf16 v8bf;
typedef __attribute__((ext_vector_type(8)))  float  v8f;

__device__ __forceinline__ v8f wmma_bf16(v16bf a, v16bf b, v8f c) {
  // D = A(16x32) * B(32x16) + C, f32 accumulate
  return __builtin_amdgcn_wmma_f32_16x16x32_bf16(
      /*neg_a=*/false, a, /*neg_b=*/false, b,
      /*c_mod=*/(short)0, c, /*reuse_a=*/false, /*reuse_b=*/false);
}

// A fragment: 16x32 (MxK) bf16, row-major LDS tile [rows][128]
// lane L: m = m0 + (L&15); elements 0..7 -> k = k0..k0+7, 8..15 -> k0+16..k0+23
// with k0 = kk*32 + 8*(L>=16)
__device__ __forceinline__ v16bf load_frag_A(const __bf16* tile, int m0, int kk, int lane) {
  int m  = m0 + (lane & 15);
  int k0 = kk * 32 + ((lane >> 4) ? 8 : 0);
  const __bf16* p = tile + m * D + k0;
  v8bf lo = *(const v8bf*)(p);
  v8bf hi = *(const v8bf*)(p + 16);
  v16bf f;
#pragma unroll
  for (int e = 0; e < 8; ++e) { f[e] = lo[e]; f[8 + e] = hi[e]; }
  return f;
}

// B fragment: 32x16 (KxN) where B[k][n] = W[n][k] (W row-major [128][128] bf16)
// lane L: n = c0 + (L&15); elements 0..7 -> k = k0..k0+7, 8..15 -> k0+16..k0+23, k0 = kk*32 + 8*(L>=16)
__device__ __forceinline__ v16bf load_frag_B(const __bf16* w, int c0, int kk, int lane) {
  int nn = c0 + (lane & 15);
  int k0 = kk * 32 + ((lane >> 4) ? 8 : 0);
  const __bf16* p = w + nn * D + k0;
  v8bf lo = *(const v8bf*)(p);
  v8bf hi = *(const v8bf*)(p + 16);
  v16bf f;
#pragma unroll
  for (int e = 0; e < 8; ++e) { f[e] = lo[e]; f[8 + e] = hi[e]; }
  return f;
}

// Global f32 -> LDS bf16 staging (coalesced float4 reads)
__device__ __forceinline__ void load_g2l_bf16(const float* __restrict__ g, __bf16* l,
                                              int n4, int t, int nt) {
  const float4* g4 = (const float4*)g;
  for (int i = t; i < n4; i += nt) {
    float4 v = g4[i];
    __bf16* p = l + 4 * i;
    p[0] = (__bf16)v.x; p[1] = (__bf16)v.y; p[2] = (__bf16)v.z; p[3] = (__bf16)v.w;
  }
}

// tc / tv per row of this chunk
__device__ __forceinline__ void compute_tctv(const float* __restrict__ taskc,
                                             const float* __restrict__ taskv,
                                             int b, int r0, float* tcL, float* tvL, int t) {
  int r = r0 + t;
  float tc = (r == 0) ? (0.01f * MAX_NUM) : taskc[b * NS + (r - 1)] * 0.01f;
  float tv = (r == 0) ? 0.0f : taskv[b * NS + (r - 1)];
  tv = (tc < 0.01f * MAX_NUM) ? tv : 0.0f;
  tcL[t] = tc;
  tvL[t] = tv;
}

// Dual projection with shared A fragments.  Outputs are COLUMN-MAJOR in LDS:
//   outX[col * CHUNK + row]  so each lane's 8 accumulator rows (contiguous in
//   row) pack into a single 16-byte ds_store_b128.
__device__ __forceinline__ void gemm_proj2(const __bf16* tile,
                                           const __bf16* w1, const __bf16* w2,
                                           __bf16* out1, __bf16* out2,
                                           const float* tcL, const float* tvL,
                                           const float* wkd1, const float* wkd2,
                                           int wave, int lane) {
#pragma unroll
  for (int rb = 0; rb < 2; ++rb) {
    int m0 = wave * 32 + rb * 16;
    v16bf a[4];
#pragma unroll
    for (int kk = 0; kk < 4; ++kk) a[kk] = load_frag_A(tile, m0, kk, lane);
    int rbase = m0 + ((lane >> 4) ? 8 : 0);
#pragma unroll
    for (int c = 0; c < 8; ++c) {
      int col = c * 16 + (lane & 15);
      v8f acc1 = {}, acc2 = {};
#pragma unroll
      for (int kk = 0; kk < 4; ++kk)
        acc1 = wmma_bf16(a[kk], load_frag_B(w1, c * 16, kk, lane), acc1);
#pragma unroll
      for (int kk = 0; kk < 4; ++kk)
        acc2 = wmma_bf16(a[kk], load_frag_B(w2, c * 16, kk, lane), acc2);
      float w10 = wkd1[col * 2 + 0], w11 = wkd1[col * 2 + 1];
      float w20 = wkd2[col * 2 + 0], w21 = wkd2[col * 2 + 1];
      v8bf o1, o2;
#pragma unroll
      for (int jj = 0; jj < 8; ++jj) {
        int row = rbase + jj;
        o1[jj] = (__bf16)(acc1[jj] + tcL[row] * w10 + tvL[row] * w11);
        o2[jj] = (__bf16)(acc2[jj] + tcL[row] * w20 + tvL[row] * w21);
      }
      *(v8bf*)(out1 + col * CHUNK + rbase) = o1;
      *(v8bf*)(out2 + col * CHUNK + rbase) = o2;
    }
  }
}

// Single projection (kernel D), column-major output
__device__ __forceinline__ void gemm_proj1(const __bf16* tile, const __bf16* w,
                                           __bf16* outL,
                                           const float* tcL, const float* tvL,
                                           const float* wkdL, int wave, int lane) {
#pragma unroll
  for (int rb = 0; rb < 2; ++rb) {
    int m0 = wave * 32 + rb * 16;
    v16bf a[4];
#pragma unroll
    for (int kk = 0; kk < 4; ++kk) a[kk] = load_frag_A(tile, m0, kk, lane);
    int rbase = m0 + ((lane >> 4) ? 8 : 0);
#pragma unroll
    for (int c = 0; c < 8; ++c) {
      int col = c * 16 + (lane & 15);
      v8f acc = {};
#pragma unroll
      for (int kk = 0; kk < 4; ++kk)
        acc = wmma_bf16(a[kk], load_frag_B(w, c * 16, kk, lane), acc);
      float wk0 = wkdL[col * 2 + 0], wk1 = wkdL[col * 2 + 1];
      v8bf o;
#pragma unroll
      for (int jj = 0; jj < 8; ++jj) {
        int row = rbase + jj;
        o[jj] = (__bf16)(acc[jj] + tcL[row] * wk0 + tvL[row] * wk1);
      }
      *(v8bf*)(outL + col * CHUNK + rbase) = o;
    }
  }
}

// ---------------------------------------------------------------------------
// Kernel 1: Q1 = graph_embedding @ Wq_fixed^T + step_context @ Wq_step^T
// ---------------------------------------------------------------------------
__global__ void q_kernel(const float* __restrict__ gemb, const float* __restrict__ stepc,
                         const float* __restrict__ Wqf, const float* __restrict__ Wqs,
                         float* __restrict__ ws) {
  int b = blockIdx.x;
  int c = threadIdx.x;   // 128 threads
  const float* ge = gemb + b * D;
  const float* sc = stepc + b * (2 * D + 32);
  float acc = 0.f;
  for (int k = 0; k < D; ++k) acc += ge[k] * Wqf[c * D + k];
  for (int k = 0; k < 2 * D + 32; ++k) acc += sc[k] * Wqs[c * (2 * D + 32) + k];
  ws[OFF_Q1 + b * D + c] = acc;
}

// ---------------------------------------------------------------------------
// Kernel 2: per (chunk, batch): K1/V via WMMA, masked-softmax partials,
//           score min/max partials.  256 threads, ~270KB dynamic LDS.
//           Only 1 workgroup fits per WGP (LDS-bound) -> give the register
//           allocator the whole VGPR file (no spills).
// ---------------------------------------------------------------------------
__global__ void __launch_bounds__(256, 1)
proj_att_kernel(const float* __restrict__ node,
                const float* __restrict__ taskc,
                const float* __restrict__ taskv,
                const unsigned char* __restrict__ mask,
                const float* __restrict__ Wks1,
                const float* __restrict__ Wvs,
                const float* __restrict__ Wkd1,
                const float* __restrict__ Wvd,
                float* __restrict__ ws) {
  extern __shared__ char smem[];
  __bf16* nodeL = (__bf16*)(smem);               // 32768 elems (64KB), row-major
  __bf16* w1L   = (__bf16*)(smem + 65536);       // 16384 (32KB)
  __bf16* w2L   = (__bf16*)(smem + 98304);       // 16384
  __bf16* k1L   = (__bf16*)(smem + 131072);      // 32768, COLUMN-major [128][256]
  __bf16* vL    = (__bf16*)(smem + 196608);      // 32768, COLUMN-major [128][256]
  float*  tcL   = (float*)(smem + 262144);       // 256
  float*  tvL   = (float*)(smem + 263168);       // 256
  float*  q1L   = (float*)(smem + 264192);       // 128
  float*  wkd1L = (float*)(smem + 264704);       // 256
  float*  wvdL  = (float*)(smem + 265728);       // 256
  float*  attL  = (float*)(smem + 266752);       // 2048 ([8][256])
  float*  redL  = (float*)(smem + 274944);       // 256

  int j = blockIdx.x;          // chunk
  int b = blockIdx.y;          // batch
  int t = threadIdx.x;         // 0..255
  int wave = t >> 5, lane = t & 31;
  int r0 = j * CHUNK;

  // --- stage inputs ---
  load_g2l_bf16(node + (size_t)(b * N + r0) * D, nodeL, CHUNK * D / 4, t, 256);
  load_g2l_bf16(Wks1, w1L, D * D / 4, t, 256);
  load_g2l_bf16(Wvs,  w2L, D * D / 4, t, 256);
  if (t < 256) { wkd1L[t] = Wkd1[t]; wvdL[t] = Wvd[t]; }
  if (t < D) q1L[t] = ws[OFF_Q1 + b * D + t];
  compute_tctv(taskc, taskv, b, r0, tcL, tvL, t);
  bool masked = mask[b * N + r0 + t] != 0;
  __syncthreads();

  // --- score min/max partials for this chunk ---
  float scr = fmaxf(tvL[t], 0.f) / (fmaxf(tcL[t], 0.f) + TINYV);
  if (masked) scr = 0.f;
  scr = fminf(scr, 5.0f);
  redL[t] = scr; __syncthreads();
  for (int s = 128; s > 0; s >>= 1) { if (t < s) redL[t] = fmaxf(redL[t], redL[t + s]); __syncthreads(); }
  if (t == 0) ws[OFF_MM + (b * NCHUNK + j) * 2 + 1] = redL[0];
  __syncthreads();
  redL[t] = scr; __syncthreads();
  for (int s = 128; s > 0; s >>= 1) { if (t < s) redL[t] = fminf(redL[t], redL[t + s]); __syncthreads(); }
  if (t == 0) ws[OFF_MM + (b * NCHUNK + j) * 2 + 0] = redL[0];
  __syncthreads();

  // --- WMMA projections: K1 and V (shared A fragments, col-major outputs) ---
  gemm_proj2(nodeL, w1L, w2L, k1L, vL, tcL, tvL, wkd1L, wvdL, wave, lane);
  __syncthreads();

  // --- attention scores: att[h][row] = (Qh . K1h_row)/4, masked ---
#pragma unroll
  for (int h = 0; h < NHEADS; ++h) {
    float a = 0.f;
#pragma unroll
    for (int e = 0; e < DH; ++e)
      a += q1L[h * DH + e] * (float)k1L[(h * DH + e) * CHUNK + t];
    a *= INV_SQRT_DH;
    if (masked) a = NEG_INF;
    attL[h * CHUNK + t] = a;
  }
  __syncthreads();

  // --- per-head streaming-softmax partials (warp w handles head w) ---
  {
    const float* av = attL + wave * CHUNK;
    // chunk max
    float m = -3.4e38f;
    for (int i = lane; i < CHUNK; i += 32) m = fmaxf(m, av[i]);
#pragma unroll
    for (int off = 16; off > 0; off >>= 1) m = fmaxf(m, __shfl_xor(m, off, 32));
    // sum of exp
    float s = 0.f;
    for (int i = lane; i < CHUNK; i += 32) s += expf(av[i] - m);
#pragma unroll
    for (int off = 16; off > 0; off >>= 1) s += __shfl_xor(s, off, 32);
    // acc[e] = sum_i exp(att_i - m) * V[i][h*16+e]   (V is col-major)
    int e = lane & 15, half = lane >> 4;
    const __bf16* vcol = vL + (wave * DH + e) * CHUNK;
    float a = 0.f;
    for (int i = half; i < CHUNK; i += 2)
      a += expf(av[i] - m) * (float)vcol[i];
    a += __shfl_xor(a, 16, 32);
    float* P = ws + OFF_ATTP + (size_t)((b * NCHUNK + j) * NHEADS + wave) * 18;
    if (lane == 0) { P[0] = m; P[1] = s; }
    if (lane < 16) P[2 + e] = a;
  }
}

// ---------------------------------------------------------------------------
// Kernel 3: per batch: combine partials -> Q2 -> Q2 @ Wout^T; score norm params
// ---------------------------------------------------------------------------
__global__ void combine_kernel(const float* __restrict__ Wout, float* __restrict__ ws) {
  __shared__ float q2s[D];
  int b = blockIdx.x;
  int t = threadIdx.x;        // 256
  int wave = t >> 5, lane = t & 31;

  if (t == 0) {
    float mn = 3.4e38f, mx = -3.4e38f;
    for (int j = 0; j < NCHUNK; ++j) {
      mn = fminf(mn, ws[OFF_MM + (b * NCHUNK + j) * 2 + 0]);
      mx = fmaxf(mx, ws[OFF_MM + (b * NCHUNK + j) * 2 + 1]);
    }
    ws[OFF_SNORM + b * 2 + 0] = mn;
    ws[OFF_SNORM + b * 2 + 1] = mx - mn;
  }

  // warp `wave` combines head `wave`
  {
    const float* P0 = ws + OFF_ATTP + (size_t)(b * NCHUNK * NHEADS) * 18;
    float M = -3.4e38f;
    for (int j = 0; j < NCHUNK; ++j) M = fmaxf(M, P0[(j * NHEADS + wave) * 18 + 0]);
    float st = 0.f;
    for (int j = 0; j < NCHUNK; ++j) {
      const float* P = P0 + (j * NHEADS + wave) * 18;
      st += P[1] * expf(P[0] - M);
    }
    if (lane < 16) {
      float a = 0.f;
      for (int j = 0; j < NCHUNK; ++j) {
        const float* P = P0 + (j * NHEADS + wave) * 18;
        a += P[2 + lane] * expf(P[0] - M);
      }
      q2s[wave * DH + lane] = a / st;
    }
  }
  __syncthreads();

  if (t < D) {
    float acc = 0.f;
    for (int k = 0; k < D; ++k) acc += q2s[k] * Wout[t * D + k];
    ws[OFF_Q2 + b * D + t] = acc;
  }
}

// ---------------------------------------------------------------------------
// Kernel 4: per (chunk, batch): recompute K2 via WMMA, logits -> unnormalized p
// ---------------------------------------------------------------------------
__global__ void __launch_bounds__(256, 1)
logits_kernel(const float* __restrict__ node,
              const float* __restrict__ taskc,
              const float* __restrict__ taskv,
              const unsigned char* __restrict__ mask,
              const float* __restrict__ Wks2,
              const float* __restrict__ Wkd2,
              float* __restrict__ ws,
              float* __restrict__ out) {
  extern __shared__ char smem[];
  __bf16* nodeL = (__bf16*)(smem);               // 64KB row-major
  __bf16* wL    = (__bf16*)(smem + 65536);       // 32KB
  __bf16* k2L   = (__bf16*)(smem + 98304);       // 64KB, COLUMN-major [128][256]
  float*  tcL   = (float*)(smem + 163840);       // 256
  float*  tvL   = (float*)(smem + 164864);       // 256
  float*  wkdL  = (float*)(smem + 165888);       // 256
  float*  q2L   = (float*)(smem + 166912);       // 128
  float*  redL  = (float*)(smem + 167424);       // 256

  int j = blockIdx.x, b = blockIdx.y, t = threadIdx.x;
  int wave = t >> 5, lane = t & 31;
  int r0 = j * CHUNK;

  load_g2l_bf16(node + (size_t)(b * N + r0) * D, nodeL, CHUNK * D / 4, t, 256);
  load_g2l_bf16(Wks2, wL, D * D / 4, t, 256);
  if (t < 256) wkdL[t] = Wkd2[t];
  if (t < D) q2L[t] = ws[OFF_Q2 + b * D + t];
  compute_tctv(taskc, taskv, b, r0, tcL, tvL, t);
  bool masked = mask[b * N + r0 + t] != 0;
  __syncthreads();

  gemm_proj1(nodeL, wL, k2L, tcL, tvL, wkdL, wave, lane);
  __syncthreads();

  // logits + probabilities (one row per thread; K2 is col-major)
  float lg = 0.f;
#pragma unroll 8
  for (int c = 0; c < D; ++c) lg += q2L[c] * (float)k2L[c * CHUNK + t];
  lg *= INV_SQRT_D;
  lg = CLIPV * tanhf(lg);
  if (masked) lg = NEG_INF;
  float e1 = expf(lg);

  float mn = ws[OFF_SNORM + b * 2 + 0];
  float dv = ws[OFF_SNORM + b * 2 + 1];
  float scr = fmaxf(tvL[t], 0.f) / (fmaxf(tcL[t], 0.f) + TINYV);
  if (masked) scr = 0.f;
  scr = fminf(scr, 5.0f);
  float sn = (fabsf(dv) <= 1e-5f) ? 1.0f : (scr - mn) / dv;
  float e2 = expf(-0.5f / (TINYV + sn * sn));

  float p = e1 * e2 * (masked ? 0.f : 1.f);
  p = fmaxf(p, 0.f);
  out[BS + (size_t)b * N + r0 + t] = p;   // unnormalized, finalized later

  // chunk partial sum of p
  redL[t] = p; __syncthreads();
  for (int s = 128; s > 0; s >>= 1) { if (t < s) redL[t] += redL[t + s]; __syncthreads(); }
  if (t == 0) ws[OFF_PSUM + b * NCHUNK + j] = redL[0];
}

// ---------------------------------------------------------------------------
// Kernel 5: per batch: normalize, log, argmax -> tours + log_ps
// ---------------------------------------------------------------------------
__global__ void finalize_kernel(const float* __restrict__ ws, float* __restrict__ out) {
  __shared__ float sS;
  __shared__ float bv[256];
  __shared__ int   bi[256];
  int b = blockIdx.x, t = threadIdx.x;
  if (t == 0) {
    float s = 0.f;
    for (int j = 0; j < NCHUNK; ++j) s += ws[OFF_PSUM + b * NCHUNK + j];
    if (s <= 0.f) s = 1.f;
    sS = s;
  }
  __syncthreads();
  float* lp = out + BS + (size_t)b * N;
  float best = -3.4e38f; int bidx = 0x7fffffff;
  for (int i = t; i < N; i += 256) {
    float p = lp[i] / sS;
    float v = logf(p <= 0.f ? 1e-10f : p);
    lp[i] = v;
    if (v > best) { best = v; bidx = i; }
  }
  bv[t] = best; bi[t] = bidx; __syncthreads();
  for (int s = 128; s > 0; s >>= 1) {
    if (t < s) {
      if (bv[t + s] > bv[t] || (bv[t + s] == bv[t] && bi[t + s] < bi[t])) {
        bv[t] = bv[t + s]; bi[t] = bi[t + s];
      }
    }
    __syncthreads();
  }
  if (t == 0) out[b] = (float)bi[0];
}

// ---------------------------------------------------------------------------
// Launch
// ---------------------------------------------------------------------------
extern "C" void kernel_launch(void* const* d_in, const int* in_sizes, int n_in,
                              void* d_out, int out_size, void* d_ws, size_t ws_size,
                              hipStream_t stream) {
  const unsigned char* mask = (const unsigned char*)d_in[0];
  const float* stepc = (const float*)d_in[1];
  const float* node  = (const float*)d_in[2];
  const float* gemb  = (const float*)d_in[3];
  // d_in[4] customers_embedding: unused by the reference forward
  const float* taskc = (const float*)d_in[5];
  const float* taskv = (const float*)d_in[6];
  const float* Wks1  = (const float*)d_in[7];
  const float* Wvs   = (const float*)d_in[8];
  const float* Wks2  = (const float*)d_in[9];
  const float* Wkd1  = (const float*)d_in[10];
  const float* Wvd   = (const float*)d_in[11];
  const float* Wkd2  = (const float*)d_in[12];
  const float* Wqf   = (const float*)d_in[13];
  const float* Wout  = (const float*)d_in[14];
  const float* Wqs   = (const float*)d_in[15];
  float* out = (float*)d_out;
  float* ws  = (float*)d_ws;

  const size_t SMEM_B = 275968;  // proj_att_kernel dynamic LDS (<= 320KB WGP)
  const size_t SMEM_D = 168448;  // logits_kernel dynamic LDS

  q_kernel<<<BS, 128, 0, stream>>>(gemb, stepc, Wqf, Wqs, ws);
  proj_att_kernel<<<dim3(NCHUNK, BS), 256, SMEM_B, stream>>>(
      node, taskc, taskv, mask, Wks1, Wvs, Wkd1, Wvd, ws);
  combine_kernel<<<BS, 256, 0, stream>>>(Wout, ws);
  logits_kernel<<<dim3(NCHUNK, BS), 256, SMEM_D, stream>>>(
      node, taskc, taskv, mask, Wks2, Wkd2, ws, out);
  finalize_kernel<<<BS, 256, 0, stream>>>(ws, out);
}
